// TGCN2_27650999451685
// MI455X (gfx1250) — matrix operand order repaired and checked by reading, not compiled, not run on hardware
//
#include <hip/hip_runtime.h>

typedef __attribute__((ext_vector_type(16))) _Float16 v16h;
typedef __attribute__((ext_vector_type(8)))  float    v8f;

#define TB 16
#define TN 10000
#define TE 160000
#define TF 64
#define HST 72   // padded LDS row stride (halves) for f16 tiles: 144B, 16B-aligned,
                 // rotates banks by 36 dwords/row to avoid 8-way conflicts

// ---------------------------------------------------------------------------
// Kernel A: init workspace (per-batch edge-index mins, degree with self-loop)
// ---------------------------------------------------------------------------
__global__ __launch_bounds__(256) void tgcn_init(int* mins, float* deg) {
    int idx = blockIdx.x * blockDim.x + threadIdx.x;
    if (idx < TB) mins[idx] = 0x7fffffff;
    if (idx < TB * TN) deg[idx] = 1.0f;   // self-loop weight 1 pre-added
}

// ---------------------------------------------------------------------------
// Kernel B: per-batch min over edge_index (B,2,E)
// ---------------------------------------------------------------------------
__global__ __launch_bounds__(256) void tgcn_min(const int* __restrict__ ei, int* mins) {
    size_t idx = (size_t)blockIdx.x * blockDim.x + threadIdx.x;
    if (idx >= (size_t)TB * 2 * TE) return;
    int b = (int)(idx / (2 * TE));
    atomicMin(&mins[b], ei[idx]);
}

// ---------------------------------------------------------------------------
// Kernel C: degree scatter  deg[b][dst] += ew
// ---------------------------------------------------------------------------
__global__ __launch_bounds__(256) void tgcn_deg(const int* __restrict__ ei,
                                                const float* __restrict__ ew,
                                                const int* __restrict__ mins,
                                                float* deg) {
    size_t idx = (size_t)blockIdx.x * blockDim.x + threadIdx.x;
    if (idx >= (size_t)TB * TE) return;
    int b = (int)(idx / TE);
    int e = (int)(idx % TE);
    int mn  = mins[b];
    int dst = ei[(size_t)b * 2 * TE + TE + e] - mn;
    atomicAdd(&deg[(size_t)b * TN + dst], ew[idx]);
}

// ---------------------------------------------------------------------------
// Kernel D: dinv = rsqrt(deg) (deg>=1 always); init agg with self-loop term
// ---------------------------------------------------------------------------
__global__ __launch_bounds__(256) void tgcn_dinv(const float* __restrict__ X,
                                                 float* deg, float* agg) {
    int idx = blockIdx.x * blockDim.x + threadIdx.x;
    if (idx >= TB * TN) return;
    float d  = deg[idx];
    float di = (d > 0.0f) ? rsqrtf(d) : 0.0f;
    deg[idx] = di;                       // reuse buffer as dinv
    agg[idx] = X[idx] * di * di;         // self-loop: x[i]*dinv[i]*1*dinv[i]
}

// ---------------------------------------------------------------------------
// Kernel E: aggregate scatter  agg[b][dst] += x[src]*dinv[src]*ew*dinv[dst]
// ---------------------------------------------------------------------------
__global__ __launch_bounds__(256) void tgcn_agg(const int* __restrict__ ei,
                                                const float* __restrict__ ew,
                                                const float* __restrict__ X,
                                                const int* __restrict__ mins,
                                                const float* __restrict__ dinv,
                                                float* agg) {
    size_t idx = (size_t)blockIdx.x * blockDim.x + threadIdx.x;
    if (idx >= (size_t)TB * TE) return;
    int b = (int)(idx / TE);
    int e = (int)(idx % TE);
    int mn  = mins[b];
    int src = ei[(size_t)b * 2 * TE + e]      - mn;
    int dst = ei[(size_t)b * 2 * TE + TE + e] - mn;
    size_t nb = (size_t)b * TN;
    float nrm = dinv[nb + src] * ew[idx] * dinv[nb + dst];
    atomicAdd(&agg[nb + dst], X[nb + src] * nrm);
}

// ---------------------------------------------------------------------------
// Kernel F: fold rank-1 GCN terms:  u_g = Wc_g @ Wl_g[:F],  v_g = bc_g @ Wl_g[:F] + bl_g
// uv layout: [u_z, v_z, u_r, v_r, u_h, v_h], each F floats
// ---------------------------------------------------------------------------
__global__ __launch_bounds__(64) void tgcn_uv(
    const float* Wc_z, const float* bc_z, const float* Wl_z, const float* bl_z,
    const float* Wc_r, const float* bc_r, const float* Wl_r, const float* bl_r,
    const float* Wc_h, const float* bc_h, const float* Wl_h, const float* bl_h,
    float* uv) {
    int j = threadIdx.x;
    if (j >= TF) return;
    const float* Wc[3] = {Wc_z, Wc_r, Wc_h};
    const float* bc[3] = {bc_z, bc_r, bc_h};
    const float* Wl[3] = {Wl_z, Wl_r, Wl_h};
    const float* bl[3] = {bl_z, bl_r, bl_h};
    for (int g = 0; g < 3; ++g) {
        float u = 0.0f, v = 0.0f;
        for (int f = 0; f < TF; ++f) {
            float w = Wl[g][f * TF + j];
            u += Wc[g][f] * w;
            v += bc[g][f] * w;
        }
        uv[(2 * g + 0) * TF + j] = u;
        uv[(2 * g + 1) * TF + j] = v + bl[g][j];
    }
}

// ---------------------------------------------------------------------------
// Kernel G: fused GRU gates with WMMA.
// Block: 128 threads = 4 wave32, one 16-row tile of one batch.
// Wave w owns output columns [16w, 16w+16).
// pre_g = agg[i]*u_g[j] + v_g[j] + (A_g @ Wl_g[F:2F])[i,j]
//   A_z = A_r = H tile;  A_h = H*r tile.
// out = z*H + (1-z)*tanh(pre_h)
//
// LDS layouts (f16 tiles use padded stride HST for bank spread + contiguous
// 128-bit fragment loads):
//   sH16 [row][k]      : A-matrix source for z/r
//   sHR  [row][k]      : A-matrix source for h (H*r)
//   sWT  [g][col][k]   : TRANSPOSED bottom-half weights -> B fragment is two
//                        contiguous 16B runs per lane = 2x ds_load_b128
// ---------------------------------------------------------------------------
__global__ __launch_bounds__(128) void tgcn_gates(
    const float* __restrict__ H,
    const float* __restrict__ Wl_z, const float* __restrict__ Wl_r,
    const float* __restrict__ Wl_h,
    const float* __restrict__ agg, const float* __restrict__ uv,
    float* __restrict__ out) {

    __shared__ _Float16 sH16[16 * HST];
    __shared__ _Float16 sHR [16 * HST];
    __shared__ _Float16 sWT [3][TF * HST];
    __shared__ float    sH32[16 * TF];
    __shared__ float    sZ  [16 * TF];
    __shared__ float    sAgg[16];

    const int b    = blockIdx.y;
    const int row0 = blockIdx.x * 16;
    const int tid  = threadIdx.x;

    // ---- cooperative staging ----
    const float* Hb = H + ((size_t)b * TN + row0) * TF;   // 16 contiguous rows
    for (int i = tid; i < 16 * TF; i += 128) {
        float v = Hb[i];
        int r = i >> 6, c = i & 63;
        sH32[i] = v;
        sH16[r * HST + c] = (_Float16)v;
    }
    // weights: read row-major (coalesced), write transposed: sWT[g][j][k]
    for (int i = tid; i < TF * TF; i += 128) {
        int k = i >> 6, j = i & 63;
        size_t off = (size_t)(TF + k) * TF + j;           // rows F..2F-1
        sWT[0][j * HST + k] = (_Float16)Wl_z[off];
        sWT[1][j * HST + k] = (_Float16)Wl_r[off];
        sWT[2][j * HST + k] = (_Float16)Wl_h[off];
    }
    if (tid < 16) sAgg[tid] = agg[(size_t)b * TN + row0 + tid];
    __syncthreads();

    const int lane = tid & 31;
    const int wave = tid >> 5;
    const int n0   = wave * 16;      // this wave's column tile
    const int mrow = lane & 15;      // A-matrix row for this lane
    const int hi   = lane >> 4;      // lane-half selector
    const int j    = n0 + (lane & 15);

    // A fragment: 16-bit A 16x32 layout (ISA 7.12.2):
    //   elems 0..7  -> K = kb*32 + hi*8 + t     (two contiguous 16B runs)
    //   elems 8..15 -> K = kb*32 + 16 + hi*8 + t
    auto loadA = [&](const _Float16* src, int kb) {
        v16h a;
        const _Float16* p = src + mrow * HST + kb * 32 + hi * 8;
        #pragma unroll
        for (int t = 0; t < 8; ++t) { a[t] = p[t]; a[8 + t] = p[16 + t]; }
        return a;
    };
    // B fragment: same K order as A (elems 0..7 -> hi*8+t, 8..15 -> 16+hi*8+t); lane->col.
    // Transposed LDS layout makes this two contiguous 16B runs per lane.
    auto loadB = [&](const _Float16* wt, int kb) {
        v16h r;
        const _Float16* p = wt + (size_t)j * HST + kb * 32 + hi * 8;
        #pragma unroll
        for (int t = 0; t < 8; ++t) { r[t] = p[t]; r[8 + t] = p[16 + t]; }
        return r;
    };

    // ---- z and r gates: (H @ Wl[F:]) via WMMA, K=64 -> 2 steps ----
    v8f accz = {}; v8f accr = {};
    #pragma unroll
    for (int kb = 0; kb < 2; ++kb) {
        v16h a  = loadA(sH16, kb);
        v16h bz = loadB(sWT[0], kb);
        accz = __builtin_amdgcn_wmma_f32_16x16x32_f16(false, a, false, bz,
                                                      (short)0, accz, false, false);
        v16h br = loadB(sWT[1], kb);
        accr = __builtin_amdgcn_wmma_f32_16x16x32_f16(false, a, false, br,
                                                      (short)0, accr, false, false);
    }

    float uz = uv[0 * TF + j], vz = uv[1 * TF + j];
    float ur = uv[2 * TF + j], vr = uv[3 * TF + j];
    #pragma unroll
    for (int v = 0; v < 8; ++v) {
        int il = v + hi * 8;                    // C/D layout: M = v + 8*(lane>=16)
        float ag = sAgg[il];
        float z  = 1.0f / (1.0f + __expf(-(accz[v] + ag * uz + vz)));
        float r  = 1.0f / (1.0f + __expf(-(accr[v] + ag * ur + vr)));
        sZ[il * TF + j]   = z;
        sHR[il * HST + j] = (_Float16)(sH32[il * TF + j] * r);
    }
    __syncthreads();

    // ---- h-tilde gate: ((H*r) @ Wl_h[F:]) ----
    v8f acch = {};
    #pragma unroll
    for (int kb = 0; kb < 2; ++kb) {
        v16h a  = loadA(sHR, kb);
        v16h bh = loadB(sWT[2], kb);
        acch = __builtin_amdgcn_wmma_f32_16x16x32_f16(false, a, false, bh,
                                                      (short)0, acch, false, false);
    }

    float uh = uv[4 * TF + j], vh = uv[5 * TF + j];
    float* Ob = out + ((size_t)b * TN + row0) * TF;
    #pragma unroll
    for (int v = 0; v < 8; ++v) {
        int il = v + hi * 8;
        float ag = sAgg[il];
        float ht = tanhf(acch[v] + ag * uh + vh);
        float z  = sZ[il * TF + j];
        Ob[il * TF + j] = z * sH32[il * TF + j] + (1.0f - z) * ht;
    }
}

// ---------------------------------------------------------------------------
extern "C" void kernel_launch(void* const* d_in, const int* in_sizes, int n_in,
                              void* d_out, int out_size, void* d_ws, size_t ws_size,
                              hipStream_t stream) {
    const float* X    = (const float*)d_in[0];
    const int*   ei   = (const int*)  d_in[1];
    const float* ew   = (const float*)d_in[2];
    const float* H    = (const float*)d_in[3];
    const float* Wc_z = (const float*)d_in[4];
    const float* bc_z = (const float*)d_in[5];
    const float* Wl_z = (const float*)d_in[6];
    const float* bl_z = (const float*)d_in[7];
    const float* Wc_r = (const float*)d_in[8];
    const float* bc_r = (const float*)d_in[9];
    const float* Wl_r = (const float*)d_in[10];
    const float* bl_r = (const float*)d_in[11];
    const float* Wc_h = (const float*)d_in[12];
    const float* bc_h = (const float*)d_in[13];
    const float* Wl_h = (const float*)d_in[14];
    const float* bl_h = (const float*)d_in[15];
    float* out = (float*)d_out;

    // workspace layout (256B-aligned sections)
    char*  ws   = (char*)d_ws;
    int*   mins = (int*)ws;                                        // B ints
    float* deg  = (float*)(ws + 256);                              // B*N  (becomes dinv)
    float* agg  = (float*)(ws + 256 + sizeof(float) * TB * TN);    // B*N
    float* uv   = (float*)(ws + 256 + 2 * sizeof(float) * TB * TN);// 6*F

    const int T = 256;
    tgcn_init<<<(TB * TN + T - 1) / T, T, 0, stream>>>(mins, deg);
    tgcn_min <<<((size_t)TB * 2 * TE + T - 1) / T, T, 0, stream>>>(ei, mins);
    tgcn_deg <<<((size_t)TB * TE + T - 1) / T, T, 0, stream>>>(ei, ew, mins, deg);
    tgcn_dinv<<<(TB * TN + T - 1) / T, T, 0, stream>>>(X, deg, agg);
    tgcn_agg <<<((size_t)TB * TE + T - 1) / T, T, 0, stream>>>(ei, ew, X, mins, deg, agg);
    tgcn_uv  <<<1, 64, 0, stream>>>(Wc_z, bc_z, Wl_z, bl_z,
                                    Wc_r, bc_r, Wl_r, bl_r,
                                    Wc_h, bc_h, Wl_h, bl_h, uv);

    dim3 grid(TN / 16, TB);           // 625 x 16 tiles
    tgcn_gates<<<grid, 128, 0, stream>>>(H, Wl_z, Wl_r, Wl_h, agg, uv, out);
}
